// SelfMultiheadAttention_45681272160319
// MI455X (gfx1250) — matrix operand order repaired and checked by reading, not compiled, not run on hardware
//
#include <hip/hip_runtime.h>
#include <hip/hip_bf16.h>

typedef __attribute__((ext_vector_type(16))) _Float16 v16h;
typedef __attribute__((ext_vector_type(8)))  _Float16 v8h;
typedef __attribute__((ext_vector_type(8)))  float    v8f;

#define WAVES_PER_BLOCK 8
#define NTHREADS (WAVES_PER_BLOCK * 32)

static constexpr int Bsz = 4, Lsz = 1024, Dsz = 768, Hn = 12, HDm = 64;

__device__ __forceinline__ int lane_id() { return (int)(threadIdx.x & 31); }
__device__ __forceinline__ int wave_id() { return (int)(threadIdx.x >> 5); }

// ---------------------------------------------------------------------------
// WMMA fragment helpers (contiguous K order: lane half h holds K = 16h..16h+15)
// ---------------------------------------------------------------------------

// A fragment: 16x32 f16, source row-major with leading dim lda (elements),
// p points at tile origin [row0, k0]. Lane m=lane&15; lanes 16-31 hold
// K=16..31; halves h -> K = kb + h.
__device__ __forceinline__ v16h load_a_frag(const _Float16* p, int lda) {
  int lane = lane_id();
  int m  = lane & 15;
  int kb = (lane >> 4) << 4;          // 0 or 16
  const _Float16* base = p + m * lda + kb;
  v16h f;
  f = *(const v16h*)(base);           // K = kb .. kb+15
  return f;
}

// B fragment: 32x16 f16 where B[k][n] = W[n][k], W row-major, leading dim ldw,
// w points at tile origin [n0, k0]. Lane n=lane&15; lanes 16-31 hold K=16..31;
// halves h -> K = kb + h.  One contiguous 32-byte load per lane.
__device__ __forceinline__ v16h load_b_fragT(const _Float16* w, int ldw) {
  int lane = lane_id();
  int n  = lane & 15;
  int kb = (lane >> 4) << 4;          // 0 or 16
  return *(const v16h*)(w + n * ldw + kb);
}

__device__ __forceinline__ v8f wmma_f16(v16h a, v16h b, v8f c) {
  return __builtin_amdgcn_wmma_f32_16x16x32_f16(
      /*neg_a=*/false, a, /*neg_b=*/false, b,
      /*c_mod=*/(short)0, c, /*reuse_a=*/false, /*reuse_b=*/false);
}

// ---------------------------------------------------------------------------
// Kernel 0: f32 -> f16 conversion
// ---------------------------------------------------------------------------
__global__ void cvt_f32_f16_kernel(const float* __restrict__ in,
                                   _Float16* __restrict__ out, int nelem) {
  int i = blockIdx.x * blockDim.x + threadIdx.x;
  if (i < nelem) out[i] = (_Float16)in[i];
}

// ---------------------------------------------------------------------------
// Kernel 1: QKV GEMM.  qkv[n,e] = sum_d x[n,d] * w_in[e,d] + b_in[e]
// Each wave: 16 rows x 64 cols (4 accumulator tiles), K-loop over 768.
// Epilogue scatters to q (scaled), k, and v-transposed buffers (f16).
// ---------------------------------------------------------------------------
__global__ void __launch_bounds__(NTHREADS)
qkv_gemm_kernel(const _Float16* __restrict__ xh,
                const _Float16* __restrict__ wh,
                const float* __restrict__ b_in,
                _Float16* __restrict__ qbuf,
                _Float16* __restrict__ kbuf,
                _Float16* __restrict__ vtbuf) {
  int wid  = blockIdx.x * WAVES_PER_BLOCK + wave_id();
  int mt   = wid & 255;                 // 256 row tiles of 16 (B*L = 4096)
  int nt   = wid >> 8;                  // 36 col tiles of 64 (3D = 2304)
  int row0 = mt * 16;
  int col0 = nt * 64;
  int lane = lane_id();
  int n    = lane & 15;
  int mb   = (lane >> 4) << 3;

  v8f acc[4] = {};
  const _Float16* a_base = xh + (size_t)row0 * Dsz;
#pragma unroll 1
  for (int k0 = 0; k0 < Dsz; k0 += 32) {
    __builtin_prefetch(a_base + k0 + 64, 0, 1);
    v16h af = load_a_frag(a_base + k0, Dsz);
#pragma unroll
    for (int j = 0; j < 4; ++j) {
      v16h bf = load_b_fragT(wh + (size_t)(col0 + j * 16) * Dsz + k0, Dsz);
      acc[j] = wmma_f16(af, bf, acc[j]);
    }
  }

  int b = row0 >> 10;                   // tile never crosses a batch boundary
#pragma unroll
  for (int j = 0; j < 4; ++j) {
    int e = col0 + j * 16 + n;
    float bias = b_in[e];
#pragma unroll
    for (int r = 0; r < 8; ++r) {
      int row = row0 + mb + r;
      int l   = row & (Lsz - 1);
      float val = acc[j][r] + bias;
      if (e < Dsz) {                               // Q, pre-scaled by HD^-0.5
        int h = e >> 6, hd = e & 63;
        qbuf[(((size_t)(b * Hn + h) * Lsz) + l) * HDm + hd] =
            (_Float16)(val * 0.125f);
      } else if (e < 2 * Dsz) {                    // K
        int e2 = e - Dsz; int h = e2 >> 6, hd = e2 & 63;
        kbuf[(((size_t)(b * Hn + h) * Lsz) + l) * HDm + hd] = (_Float16)val;
      } else {                                     // V transposed: [B,H,HD,L]
        int e2 = e - 2 * Dsz; int h = e2 >> 6, hd = e2 & 63;
        vtbuf[(((size_t)(b * Hn + h) * HDm) + hd) * Lsz + l] = (_Float16)val;
      }
    }
  }
}

// ---------------------------------------------------------------------------
// Kernel 2: flash-style attention with bias.  One wave per (b,h,16 q-rows).
// S = Q K^T + bias, online softmax, O += P V (via LDS C->A layout transpose).
// The 201 MB f32 attn_bias stream is the roofline-binding resource; next-chunk
// bias/K/V tiles are prefetched to keep HBM saturated while WMMAs/softmax run.
// ---------------------------------------------------------------------------
__global__ void __launch_bounds__(NTHREADS)
attn_kernel(const _Float16* __restrict__ qbuf,
            const _Float16* __restrict__ kbuf,
            const _Float16* __restrict__ vtbuf,
            const float* __restrict__ attn_bias,
            _Float16* __restrict__ obuf) {
  __shared__ _Float16 lds_p[WAVES_PER_BLOCK][16 * 32];

  int wid = blockIdx.x * WAVES_PER_BLOCK + wave_id();
  int qt  = wid & 63;                    // 64 q tiles of 16
  int bh  = wid >> 6;                    // 48 (b,h) pairs
  int b   = bh / Hn, h = bh % Hn;
  int lane = lane_id();
  int n    = lane & 15;
  int mb   = (lane >> 4) << 3;
  _Float16* lp = lds_p[wave_id()];

  const _Float16* qbh  = qbuf  + (size_t)bh * Lsz * HDm + (size_t)qt * 16 * HDm;
  const _Float16* kbh  = kbuf  + (size_t)bh * Lsz * HDm;
  const _Float16* vbh  = vtbuf + (size_t)bh * HDm * Lsz;
  const float*    bias = attn_bias + (size_t)bh * Lsz * Lsz + (size_t)qt * 16 * Lsz;

  // Q fragments for K=0..31 and K=32..63 (loaded once, reused 32x each)
  v16h qf0 = load_a_frag(qbh,      HDm);
  v16h qf1 = load_a_frag(qbh + 32, HDm);

  v8f accO[4] = {};
  float m_i[8], l_i[8];
#pragma unroll
  for (int r = 0; r < 8; ++r) { m_i[r] = -1e30f; l_i[r] = 0.0f; }

#pragma unroll 1
  for (int key0 = 0; key0 < Lsz; key0 += 32) {
    // prefetch next chunk's K rows and V columns (hides HBM latency)
    __builtin_prefetch(kbh + (size_t)(key0 + 32) * HDm + n * HDm, 0, 1);
    __builtin_prefetch(vbh + (size_t)n * Lsz + key0 + 32, 0, 1);

    // S tiles: rows = q, cols = 32 keys (two 16-wide WMMA tiles), K = HD = 64
    v8f s0 = {}, s1 = {};
    s0 = wmma_f16(qf0, load_b_fragT(kbh + (size_t)key0 * HDm,             HDm), s0);
    s0 = wmma_f16(qf1, load_b_fragT(kbh + (size_t)key0 * HDm + 32,        HDm), s0);
    s1 = wmma_f16(qf0, load_b_fragT(kbh + (size_t)(key0 + 16) * HDm,      HDm), s1);
    s1 = wmma_f16(qf1, load_b_fragT(kbh + (size_t)(key0 + 16) * HDm + 32, HDm), s1);

    // bias add + online softmax; row M = mb+r lives in lanes (n, same half)
#pragma unroll
    for (int r = 0; r < 8; ++r) {
      const float* bp = bias + (size_t)(mb + r) * Lsz + key0 + n;
      __builtin_prefetch(bp + 32, 0, 1);   // next key-chunk, same bias row
      float t0 = s0[r] + bp[0];
      float t1 = s1[r] + bp[16];
      float t = fmaxf(t0, t1);
      t = fmaxf(t, __shfl_xor(t, 1, 32));
      t = fmaxf(t, __shfl_xor(t, 2, 32));
      t = fmaxf(t, __shfl_xor(t, 4, 32));
      t = fmaxf(t, __shfl_xor(t, 8, 32));
      float nm = fmaxf(m_i[r], t);
      float sc = __expf(m_i[r] - nm);
      float p0 = __expf(t0 - nm);
      float p1 = __expf(t1 - nm);
      float rs = p0 + p1;
      rs += __shfl_xor(rs, 1, 32);
      rs += __shfl_xor(rs, 2, 32);
      rs += __shfl_xor(rs, 4, 32);
      rs += __shfl_xor(rs, 8, 32);
      l_i[r] = l_i[r] * sc + rs;
      m_i[r] = nm;
      accO[0][r] *= sc; accO[1][r] *= sc; accO[2][r] *= sc; accO[3][r] *= sc;
      // P tile to LDS in [m][k] order for the A-fragment reload
      lp[(mb + r) * 32 + n]      = (_Float16)p0;
      lp[(mb + r) * 32 + 16 + n] = (_Float16)p1;
    }
    asm volatile("s_wait_dscnt 0" ::: "memory");

    v16h pf = load_a_frag(lp, 32);      // C-layout -> A-layout transpose via LDS
    // (LDS is issued/executed in-order per wave: the next iteration's
    //  ds_stores cannot pass these ds_loads, so no WAR wait is needed.)

    // O += P(16x32) * V(32x64); B from vT rows is K-contiguous
#pragma unroll
    for (int j = 0; j < 4; ++j) {
      v16h vf = load_b_fragT(vbh + (size_t)(j * 16) * Lsz + key0, Lsz);
      accO[j] = wmma_f16(pf, vf, accO[j]);
    }
  }

  // normalize and store O (f16) into [B*L, D] at columns h*64 + ...
#pragma unroll
  for (int r = 0; r < 8; ++r) {
    float inv = 1.0f / l_i[r];
    int row = b * Lsz + qt * 16 + mb + r;
#pragma unroll
    for (int j = 0; j < 4; ++j) {
      obuf[(size_t)row * Dsz + h * HDm + j * 16 + n] =
          (_Float16)(accO[j][r] * inv);
    }
  }
}

// ---------------------------------------------------------------------------
// Kernel 3: output projection.  out[n,e] = sum_d o[n,d]*w_out[e,d] + b_out[e]
// ---------------------------------------------------------------------------
__global__ void __launch_bounds__(NTHREADS)
out_gemm_kernel(const _Float16* __restrict__ oh,
                const _Float16* __restrict__ wh,
                const float* __restrict__ b_out,
                float* __restrict__ out) {
  int wid  = blockIdx.x * WAVES_PER_BLOCK + wave_id();
  int mt   = wid & 255;                 // 256 row tiles
  int nt   = wid >> 8;                  // 12 col tiles of 64
  int row0 = mt * 16;
  int col0 = nt * 64;
  int lane = lane_id();
  int n    = lane & 15;
  int mb   = (lane >> 4) << 3;

  v8f acc[4] = {};
  const _Float16* a_base = oh + (size_t)row0 * Dsz;
#pragma unroll 1
  for (int k0 = 0; k0 < Dsz; k0 += 32) {
    __builtin_prefetch(a_base + k0 + 64, 0, 1);
    v16h af = load_a_frag(a_base + k0, Dsz);
#pragma unroll
    for (int j = 0; j < 4; ++j) {
      v16h bf = load_b_fragT(wh + (size_t)(col0 + j * 16) * Dsz + k0, Dsz);
      acc[j] = wmma_f16(af, bf, acc[j]);
    }
  }
#pragma unroll
  for (int j = 0; j < 4; ++j) {
    int e = col0 + j * 16 + n;
    float bias = b_out[e];
#pragma unroll
    for (int r = 0; r < 8; ++r)
      out[(size_t)(row0 + mb + r) * Dsz + e] = acc[j][r] + bias;
  }
}

// ---------------------------------------------------------------------------
// Host launch
// ---------------------------------------------------------------------------
extern "C" void kernel_launch(void* const* d_in, const int* in_sizes, int n_in,
                              void* d_out, int out_size, void* d_ws, size_t ws_size,
                              hipStream_t stream) {
  const float* x         = (const float*)d_in[0];   // [4,1024,768]
  const float* attn_bias = (const float*)d_in[1];   // [4,12,1024,1024]
  const float* w_in      = (const float*)d_in[2];   // [2304,768]
  const float* b_in      = (const float*)d_in[3];   // [2304]
  const float* w_out     = (const float*)d_in[4];   // [768,768]
  const float* b_out     = (const float*)d_in[5];   // [768]
  float* out = (float*)d_out;                       // [4,1024,768]

  const size_t nX   = (size_t)Bsz * Lsz * Dsz;      // 3,145,728
  const size_t nWin = (size_t)3 * Dsz * Dsz;        // 1,769,472
  const size_t nWo  = (size_t)Dsz * Dsz;            //   589,824
  const size_t nQKV = (size_t)Bsz * Hn * Lsz * HDm; // 3,145,728 each

  char* ws = (char*)d_ws;
  size_t off = 0;
  auto alloc = [&](size_t elems) {
    _Float16* p = (_Float16*)(ws + off);
    off += ((elems * sizeof(_Float16)) + 255) & ~(size_t)255;
    return p;
  };
  _Float16* xh  = alloc(nX);
  _Float16* wih = alloc(nWin);
  _Float16* woh = alloc(nWo);
  _Float16* qb  = alloc(nQKV);
  _Float16* kb  = alloc(nQKV);
  _Float16* vtb = alloc(nQKV);
  _Float16* ob  = alloc(nX);
  (void)ws_size;

  // 0) f32 -> f16 conversions
  cvt_f32_f16_kernel<<<(int)((nX   + 255) / 256), 256, 0, stream>>>(x,     xh,  (int)nX);
  cvt_f32_f16_kernel<<<(int)((nWin + 255) / 256), 256, 0, stream>>>(w_in,  wih, (int)nWin);
  cvt_f32_f16_kernel<<<(int)((nWo  + 255) / 256), 256, 0, stream>>>(w_out, woh, (int)nWo);

  // 1) QKV projection: 256 row-tiles x 36 col-tiles = 9216 waves
  qkv_gemm_kernel<<<9216 / WAVES_PER_BLOCK, NTHREADS, 0, stream>>>(
      xh, wih, b_in, qb, kb, vtb);

  // 2) attention: 48 (b,h) x 64 q-tiles = 3072 waves
  attn_kernel<<<3072 / WAVES_PER_BLOCK, NTHREADS, 0, stream>>>(
      qb, kb, vtb, attn_bias, ob);

  // 3) output projection: 256 x 12 = 3072 waves
  out_gemm_kernel<<<3072 / WAVES_PER_BLOCK, NTHREADS, 0, stream>>>(
      ob, woh, b_out, out);
}
